// ResnetConstMultiSpherePointnet_63788854280405
// MI455X (gfx1250) — matrix operand order repaired and checked by reading, not compiled, not run on hardware
//
#include <hip/hip_runtime.h>

// ---------------- types ----------------
typedef __attribute__((ext_vector_type(16))) _Float16 v16h;
typedef __attribute__((ext_vector_type(8)))  _Float16 v8h;
typedef __attribute__((ext_vector_type(8)))  float    v8f;
typedef __attribute__((ext_vector_type(4)))  unsigned int u32x4;
typedef __attribute__((ext_vector_type(4)))  int i32x4v;
typedef __attribute__((ext_vector_type(8)))  int i32x8v;

#if defined(__has_builtin)
#if __has_builtin(__builtin_amdgcn_tensor_load_to_lds) && __has_builtin(__builtin_amdgcn_s_wait_tensorcnt)
#define HAVE_TDM 1
#endif
#endif

// ---------------- device helpers ----------------
__device__ __forceinline__ v16h zfrag() {
    v16h a;
#pragma unroll
    for (int j = 0; j < 16; ++j) a[j] = (_Float16)0.f;
    return a;
}

// A-fragment for V_WMMA_F32_16X16X32_F16 (16x32, f16), wave32 layout:
// lane l holds row M = l&15; K-base = 8*(l>>4); halves 0..7 -> K=base..base+7,
// halves 8..15 -> K=16+base..16+base+7.  `row` points at this lane's row, k0 folded in.
__device__ __forceinline__ v16h afrag(const _Float16* row, int kb, bool dorelu) {
    v8h lo = *(const v8h*)(row + kb);
    v8h hi = *(const v8h*)(row + kb + 16);
    if (dorelu) {
#pragma unroll
        for (int j = 0; j < 8; ++j) {
            if (lo[j] < (_Float16)0.f) lo[j] = (_Float16)0.f;
            if (hi[j] < (_Float16)0.f) hi[j] = (_Float16)0.f;
        }
    }
    return __builtin_shufflevector(lo, hi, 0,1,2,3,4,5,6,7,8,9,10,11,12,13,14,15);
}

__device__ __forceinline__ v8f wmma_f16(v16h a, v16h b, v8f c) {
    return __builtin_amdgcn_wmma_f32_16x16x32_f16(false, a, false, b, (short)0, c, false, false);
}

// order-preserving int encoding of float (0.0f -> 0), monotonic for atomicMax/Min
__device__ __forceinline__ int encf(float f) { int i = __float_as_int(f); return i >= 0 ? i : (i ^ 0x7fffffff); }
__device__ __forceinline__ float decf(int e) { int i = e >= 0 ? e : (e ^ 0x7fffffff); return __int_as_float(i); }

#ifdef HAVE_TDM
// TDM: 1-D contiguous copy of nElem 2-byte elements Global -> LDS via tensor_load_to_lds.
// D# per CDNA5 ISA: group0 {count=1, lds_addr[63:32], global_addr[120:64],
// type=2 @127:126}; group1 {data_size=1(2B)@17:16, tensor_dim0@79:48, tensor_dim1@111:80,
// tile_dim0@127:112, tile_dim1@143:128, tensor_dim0_stride@207:160}.
__device__ __forceinline__ void tdm_copy_to_lds(unsigned ldsAddr, const void* gsrc, int nElem) {
    unsigned long long ga = (unsigned long long)gsrc;
    u32x4 g0 = {};
    g0[0] = 1u;                                                  // count = 1 (valid user D#)
    g0[1] = ldsAddr;                                             // lds_addr (bytes)
    g0[2] = (unsigned)ga;                                        // global_addr[31:0]
    g0[3] = (unsigned)((ga >> 32) & 0x01ffffffu) | (2u << 30);   // global_addr[56:32], type=2
    i32x8v g1 = {};
    g1[0] = (1 << 16);                                           // data_size = 2 bytes
    g1[1] = (nElem & 0xffff) << 16;                              // tensor_dim0[15:0]
    g1[2] = (int)(((unsigned)nElem >> 16) | (1u << 16));         // tensor_dim0[31:16], tensor_dim1=1
    g1[3] = (nElem & 0xffff) << 16;                              // tile_dim0
    g1[4] = 1;                                                   // tile_dim1 = 1
    g1[5] = nElem;                                               // tensor_dim0_stride
    i32x4v z4 = {};
#if __clang_major__ >= 23
    i32x8v z8 = {};
    __builtin_amdgcn_tensor_load_to_lds(g0, g1, z4, z4, z8, 0);
#else
    __builtin_amdgcn_tensor_load_to_lds(g0, g1, z4, z4, 0);
#endif
    asm volatile("" ::: "memory");  // LDS written behind the compiler's back
}
#endif

// ---------------- weight swizzlers ----------------
// dst layout: [kt][nt][lane(32)][16 halves]; element: K = kt*32 + 8*(l>>4) + (j<8 ? j : 8+j), N = nt*16 + (l&15)
__global__ __launch_bounds__(256) void swz_mat(const float* __restrict__ src, int K, int N,
                                               _Float16* __restrict__ dst) {
    long t = (long)blockIdx.x * 256 + threadIdx.x;
    int KT = K >> 5, NT = N >> 4;
    long total = (long)KT * NT * 512;
    if (t >= total) return;
    int j = (int)(t & 15), l = (int)((t >> 4) & 31);
    long rest = t >> 9;
    int nt = (int)(rest % NT);
    int kt = (int)(rest / NT);
    int k = kt * 32 + ((l >> 4) << 3) + (j < 8 ? j : 8 + j);
    int n = nt * 16 + (l & 15);
    dst[t] = (_Float16)src[(long)k * N + n];
}

// conv OIHW -> [tap(S*S)][kt][nt][lane][16h], K = input channel (zero-padded to Ipad), N = out channel
__global__ __launch_bounds__(256) void swz_conv(const float* __restrict__ src, int O, int I, int S,
                                                int Ipad, _Float16* __restrict__ dst) {
    long t = (long)blockIdx.x * 256 + threadIdx.x;
    int KT = Ipad >> 5, NT = O >> 4;
    long total = (long)S * S * KT * NT * 512;
    if (t >= total) return;
    int j = (int)(t & 15), l = (int)((t >> 4) & 31);
    long rest = t >> 9;
    int nt = (int)(rest % NT); rest /= NT;
    int kt = (int)(rest % KT);
    int tap = (int)(rest / KT);
    int ky = tap / S, kx = tap % S;
    int k = kt * 32 + ((l >> 4) << 3) + (j < 8 ? j : 8 + j);
    int n = nt * 16 + (l & 15);
    float v = (k < I) ? src[(((long)n * I + k) * S + ky) * S + kx] : 0.f;
    dst[t] = (_Float16)v;
}

// ---------------- fc_pos: [Npts,3] @ [3,256] + b -> f16 [Npts,256] ----------------
__global__ __launch_bounds__(256) void fcpos_k(const float* __restrict__ p, const float* __restrict__ w,
                                               const float* __restrict__ b, _Float16* __restrict__ net,
                                               long npts) {
    long t = (long)blockIdx.x * 256 + threadIdx.x;
    if (t >= npts * 32) return;
    long pt = t >> 5;
    int c0 = (int)(t & 31) * 8;
    float x = p[pt * 3], y = p[pt * 3 + 1], z = p[pt * 3 + 2];
#pragma unroll
    for (int j = 0; j < 8; ++j) {
        int c = c0 + j;
        net[pt * 256 + c] = (_Float16)(x * w[c] + y * w[256 + c] + z * w[512 + c] + b[c]);
    }
}

// ---------------- ResnetBlockFC (fused): out = X@Ws + (relu(relu(X)@W0+b0))@W1 + b1 ----------------
// X (K=256) comes from x0 (K 0..127) and x1 (K 128..255).  x1_batch: x1 row = pooled[point/T].
// Weight slabs (32x128 swizzled, 8KB) are staged into LDS by the Tensor Data Mover (one
// tensor_load_to_lds per slab issued by wave 0, S_WAIT_TENSORCNT, then workgroup barrier).
__global__ __launch_bounds__(256) void resblock_wmma(
    const _Float16* __restrict__ x0, int s0,
    const _Float16* __restrict__ x1, int s1, int x1_batch, int T,
    const _Float16* __restrict__ w0s, const float* __restrict__ b0,
    const _Float16* __restrict__ w1s, const float* __restrict__ b1,
    const _Float16* __restrict__ wss,
    _Float16* __restrict__ out) {
    __shared__ alignas(32) _Float16 wbuf[2][4096];      // one 32x128 swizzled slab each
    __shared__ alignas(32) _Float16 hbuf[8][16 * 128];  // per-wave h0 tile (C->A transpose)
    const int tid = threadIdx.x, lane = tid & 31, wv = tid >> 5;
    const int m0 = blockIdx.x * 128 + wv * 16;
    const int mrow = m0 + (lane & 15);
    const int kb = (lane >> 4) << 3;
    const _Float16* r0 = x0 + (long)mrow * s0;
    const _Float16* r1 = x1_batch ? (x1 + (long)(mrow / T) * 128) : (x1 + (long)mrow * s1);

    v8f acc0[8], accS[8];
#pragma unroll
    for (int i = 0; i < 8; ++i) { v8f z = {}; acc0[i] = z; accS[i] = z; }

    // fc0 (relu input) + shortcut (raw input), fused over K=256
    for (int kt = 0; kt < 8; ++kt) {
        __syncthreads();
#ifdef HAVE_TDM
        if (tid == 0) {
            tdm_copy_to_lds((unsigned)(unsigned long long)(const void*)&wbuf[0][0],
                            w0s + (long)kt * 4096, 4096);
            tdm_copy_to_lds((unsigned)(unsigned long long)(const void*)&wbuf[1][0],
                            wss + (long)kt * 4096, 4096);
            __builtin_amdgcn_s_wait_tensorcnt(0);
        }
#else
        *(v16h*)&wbuf[0][tid * 16] = *(const v16h*)(w0s + (long)kt * 4096 + tid * 16);
        *(v16h*)&wbuf[1][tid * 16] = *(const v16h*)(wss + (long)kt * 4096 + tid * 16);
#endif
        if (kt < 7) {
            __builtin_prefetch(w0s + (long)(kt + 1) * 4096 + tid * 16, 0, 0);
            __builtin_prefetch(wss + (long)(kt + 1) * 4096 + tid * 16, 0, 0);
        }
        __syncthreads();
        const _Float16* row = (kt < 4) ? (r0 + kt * 32) : (r1 + (kt - 4) * 32);
        v16h ar = afrag(row, kb, true);
        v16h ax = afrag(row, kb, false);
#pragma unroll
        for (int nt = 0; nt < 8; ++nt) {
            v16h b0v = *(const v16h*)&wbuf[0][(nt * 32 + lane) * 16];
            v16h bsv = *(const v16h*)&wbuf[1][(nt * 32 + lane) * 16];
            acc0[nt] = wmma_f16(ar, b0v, acc0[nt]);
            accS[nt] = wmma_f16(ax, bsv, accS[nt]);
        }
    }

    // h0 = relu(acc0 + b0) -> LDS (C layout -> row-major) for fc1 A-frags
    const int mb = (lane >> 4) * 8, nn = lane & 15;
#pragma unroll
    for (int nt = 0; nt < 8; ++nt) {
        float bb = b0[nt * 16 + nn];
#pragma unroll
        for (int r = 0; r < 8; ++r) {
            float v = acc0[nt][r] + bb;
            v = v > 0.f ? v : 0.f;
            hbuf[wv][(mb + r) * 128 + nt * 16 + nn] = (_Float16)v;
        }
    }

    // fc1 over K=128, accumulated on top of the shortcut
    for (int kt = 0; kt < 4; ++kt) {
        __syncthreads();
#ifdef HAVE_TDM
        if (tid == 0) {
            tdm_copy_to_lds((unsigned)(unsigned long long)(const void*)&wbuf[0][0],
                            w1s + (long)kt * 4096, 4096);
            __builtin_amdgcn_s_wait_tensorcnt(0);
        }
#else
        *(v16h*)&wbuf[0][tid * 16] = *(const v16h*)(w1s + (long)kt * 4096 + tid * 16);
#endif
        __syncthreads();
        v16h a = afrag(&hbuf[wv][(lane & 15) * 128 + kt * 32], kb, false);
#pragma unroll
        for (int nt = 0; nt < 8; ++nt) {
            v16h bv = *(const v16h*)&wbuf[0][(nt * 32 + lane) * 16];
            accS[nt] = wmma_f16(a, bv, accS[nt]);
        }
    }

#pragma unroll
    for (int nt = 0; nt < 8; ++nt) {
        float bb = b1[nt * 16 + nn];
#pragma unroll
        for (int r = 0; r < 8; ++r)
            out[(long)(m0 + mb + r) * 128 + nt * 16 + nn] = (_Float16)(accS[nt][r] + bb);
    }
}

// ---------------- column max over T: pooled[b*128+c] ----------------
__global__ __launch_bounds__(256) void poolmax_k(const _Float16* __restrict__ net,
                                                 _Float16* __restrict__ pooled, int T) {
    __shared__ float red[256];
    int b = blockIdx.x >> 7, c = blockIdx.x & 127;
    float m = -3.0e38f;
    for (int t = threadIdx.x; t < T; t += 256)
        m = fmaxf(m, (float)net[((long)b * T + t) * 128 + c]);
    red[threadIdx.x] = m;
    __syncthreads();
    for (int s = 128; s > 0; s >>= 1) {
        if (threadIdx.x < s) red[threadIdx.x] = fmaxf(red[threadIdx.x], red[threadIdx.x + s]);
        __syncthreads();
    }
    if (threadIdx.x == 0) pooled[blockIdx.x] = (_Float16)red[0];
}

// ---------------- spherical scatter with atomics ----------------
__global__ __launch_bounds__(256) void scatter_k(
    const float* __restrict__ p, const _Float16* __restrict__ net, int T, int B,
    int* __restrict__ fmaxb, int* __restrict__ dminb, int* __restrict__ dmaxb,
    float* __restrict__ dsumb, float* __restrict__ cntb) {
    long t = (long)blockIdx.x * 256 + threadIdx.x;
    long total = (long)B * T * 4;
    if (t >= total) return;
    int l = (int)(t & 3);
    long pt = t >> 2;
    int b = (int)(pt / T);
    float x = p[pt * 3], y = p[pt * 3 + 1], z = p[pt * 3 + 2];
    const float PI_F = 3.1415927410125732f;
    float cx = ((l == 0) || (l == 3)) ? 0.25f : -0.25f;
    float cy = ((l == 0) || (l == 2)) ? 0.25f : -0.25f;
    float cz = (l == 1) ? -0.25f : 0.25f;
    float xv = x - cx, yv = y - cy, zv = z - cz;
    float dist = sqrtf(x * x + y * y + z * z);
    float lat = 90.f - atan2f(zv, sqrtf(xv * xv + yv * yv)) * (180.f / PI_F);
    float mer = fmodf(360.f + atan2f(yv, xv) * (180.f / PI_F), 360.f);
    int yg = (int)floorf(lat / (180.f / 64.f));
    int xg = (int)floorf(mer / (360.f / 64.f));
    int idx = xg + 64 * yg;
    idx = idx < 0 ? 0 : (idx > 4095 ? 4095 : idx);
    long cell = ((long)b * 4 + l) * 4096 + idx;
    atomicMin(&dminb[cell], encf(dist));
    atomicMax(&dmaxb[cell], encf(dist));
    atomicAdd(&dsumb[cell], dist);
    atomicAdd(&cntb[cell], 1.0f);
    const _Float16* f = net + pt * 128;
    int* fm = fmaxb + cell * 128;
    for (int c = 0; c < 128; ++c) atomicMax(&fm[c], encf((float)f[c]));
}

// ---------------- build UNet input x0 (NHWC f16, C padded 131 -> 160) ----------------
__global__ __launch_bounds__(256) void finalize_k(const int* __restrict__ fmaxb, const int* __restrict__ dminb,
                                                  const int* __restrict__ dmaxb, const float* __restrict__ dsumb,
                                                  const float* __restrict__ cntb, _Float16* __restrict__ x0) {
    long t = (long)blockIdx.x * 256 + threadIdx.x;
    long total = (long)65536 * 160;
    if (t >= total) return;
    int c = (int)(t % 160);
    long cell = t / 160;
    float v;
    if (c < 128) v = decf(fmaxb[cell * 128 + c]);
    else if (c == 128) v = decf(dminb[cell]);
    else if (c == 129) v = dsumb[cell] / fmaxf(cntb[cell], 1.f);
    else if (c == 130) v = decf(dmaxb[cell]);
    else v = 0.f;
    x0[cell * 160 + c] = (_Float16)v;
}

// ---------------- 3x3 conv (implicit GEMM, NHWC f16, SAME padding) ----------------
__global__ __launch_bounds__(256) void conv3x3_wmma(
    const _Float16* __restrict__ in, int H, int W, int inStride, int Cin,
    const _Float16* __restrict__ ws, const float* __restrict__ bias,
    _Float16* __restrict__ out, int outStride, int outOff, int Cout, int N, int dorelu) {
    const int lane = threadIdx.x & 31, wv = threadIdx.x >> 5;
    const int ntT = Cout >> 4, xB = W >> 4, KT = Cin >> 5;
    long tile = (long)blockIdx.x * 8 + wv;
    long total = (long)N * H * xB * ntT;
    if (tile >= total) return;
    int ntile = (int)(tile % ntT); long t2 = tile / ntT;
    int xb = (int)(t2 % xB); t2 /= xB;
    int y = (int)(t2 % H);
    int n = (int)(t2 / H);
    const int kb = (lane >> 4) << 3;
    const int x = xb * 16 + (lane & 15);
    const _Float16* img = in + (long)n * H * W * inStride;
    v8f acc = {};
    for (int ky = 0; ky < 3; ++ky) {
        const int yy = y + ky - 1;
        for (int kx = 0; kx < 3; ++kx) {
            const int xx = x + kx - 1;
            const bool oob = (yy < 0) | (yy >= H) | (xx < 0) | (xx >= W);
            const _Float16* px = img + ((long)yy * W + xx) * inStride;
            const _Float16* wt = ws + (long)((ky * 3 + kx) * KT) * ntT * 512;
            for (int kt = 0; kt < KT; ++kt) {
                v16h a = oob ? zfrag() : afrag(px + kt * 32, kb, false);
                v16h b = *(const v16h*)(wt + ((long)kt * ntT + ntile) * 512 + lane * 16);
                acc = wmma_f16(a, b, acc);
            }
        }
    }
    const int mb = (lane >> 4) * 8, nn = lane & 15;
    const float bb = bias[ntile * 16 + nn];
#pragma unroll
    for (int r = 0; r < 8; ++r) {
        float v = acc[r] + bb;
        if (dorelu && v < 0.f) v = 0.f;
        out[(((long)n * H + y) * W + (xb * 16 + mb + r)) * outStride + outOff + ntile * 16 + nn] = (_Float16)v;
    }
}

// ---------------- 1x1 conv (optional nearest-up x2, optional f32 output) ----------------
__global__ __launch_bounds__(256) void conv1x1_wmma(
    const _Float16* __restrict__ in, int Hin, int Win, int inStride, int Cin,
    const _Float16* __restrict__ ws, const float* __restrict__ bias,
    _Float16* __restrict__ outh, float* __restrict__ outf,
    int Hout, int Wout, int outStride, int Cout, int N, int dorelu, int up) {
    const int lane = threadIdx.x & 31, wv = threadIdx.x >> 5;
    const int ntT = Cout >> 4, xB = Wout >> 4, KT = Cin >> 5;
    long tile = (long)blockIdx.x * 8 + wv;
    long total = (long)N * Hout * xB * ntT;
    if (tile >= total) return;
    int ntile = (int)(tile % ntT); long t2 = tile / ntT;
    int xb = (int)(t2 % xB); t2 /= xB;
    int y = (int)(t2 % Hout);
    int n = (int)(t2 / Hout);
    const int kb = (lane >> 4) << 3;
    const int x = xb * 16 + (lane & 15);
    const int xi = up ? (x >> 1) : x, yi = up ? (y >> 1) : y;
    const _Float16* px = in + (((long)n * Hin + yi) * Win + xi) * inStride;
    v8f acc = {};
    for (int kt = 0; kt < KT; ++kt) {
        v16h a = afrag(px + kt * 32, kb, false);
        v16h b = *(const v16h*)(ws + ((long)kt * ntT + ntile) * 512 + lane * 16);
        acc = wmma_f16(a, b, acc);
    }
    const int mb = (lane >> 4) * 8, nn = lane & 15;
    const float bb = bias[ntile * 16 + nn];
#pragma unroll
    for (int r = 0; r < 8; ++r) {
        float v = acc[r] + bb;
        if (dorelu && v < 0.f) v = 0.f;
        long oi = (((long)n * Hout + y) * Wout + (xb * 16 + mb + r)) * outStride + ntile * 16 + nn;
        if (outf) outf[oi] = v;
        else outh[oi] = (_Float16)v;
    }
}

// ---------------- 2x2 max pool (NHWC f16) ----------------
__global__ __launch_bounds__(256) void maxpool2_k(const _Float16* __restrict__ in, int Hin, int Win,
                                                  int inStride, int inOff, int C,
                                                  _Float16* __restrict__ out, int outStride, int N) {
    long t = (long)blockIdx.x * 256 + threadIdx.x;
    int Ho = Hin >> 1, Wo = Win >> 1;
    long total = (long)N * Ho * Wo * C;
    if (t >= total) return;
    int c = (int)(t % C); long r = t / C;
    int xo = (int)(r % Wo); r /= Wo;
    int yo = (int)(r % Ho);
    int n = (int)(r / Ho);
    const _Float16* base = in + (((long)n * Hin + yo * 2) * Win + xo * 2) * inStride + inOff + c;
    float a = (float)base[0], b = (float)base[inStride];
    float d = (float)base[(long)Win * inStride], e = (float)base[(long)Win * inStride + inStride];
    out[(((long)n * Ho + yo) * Wo + xo) * outStride + c] = (_Float16)fmaxf(fmaxf(a, b), fmaxf(d, e));
}

__global__ void centers_k(float* __restrict__ out) {
    int t = threadIdx.x;
    if (t >= 48) return;
    const float c[12] = {0.25f, 0.25f, 0.25f, -0.25f, -0.25f, -0.25f,
                         -0.25f, 0.25f, 0.25f, 0.25f, -0.25f, 0.25f};
    out[t] = c[t % 12];
}

// ---------------- host ----------------
static inline int nb256(long n) { return (int)((n + 255) / 256); }
static inline int cgrid(long tiles) { return (int)((tiles + 7) / 8); }

extern "C" void kernel_launch(void* const* d_in, const int* in_sizes, int n_in,
                              void* d_out, int out_size, void* d_ws, size_t ws_size,
                              hipStream_t stream) {
    (void)in_sizes; (void)n_in; (void)out_size; (void)ws_size;
    const int B = 4, T = 32768;
    const long NPTS = (long)B * T;

    const float* p   = (const float*)d_in[0];
    const float* fcw = (const float*)d_in[1];
    const float* fcb = (const float*)d_in[2];

    char* base = (char*)d_ws;
    size_t off = 0;
    auto alloc = [&](size_t bytes) -> char* {
        char* r = base + off;
        off = (off + bytes + 255) & ~(size_t)255;
        return r;
    };

    _Float16* net256 = (_Float16*)alloc(NPTS * 256 * 2);
    _Float16* netA   = (_Float16*)alloc(NPTS * 128 * 2);
    _Float16* netB   = (_Float16*)alloc(NPTS * 128 * 2);
    _Float16* pooled = (_Float16*)alloc(1024);

    _Float16 *blkW0[5], *blkW1[5], *blkWS[5];
    for (int i = 0; i < 5; ++i) {
        blkW0[i] = (_Float16*)alloc(32768 * 2);
        blkW1[i] = (_Float16*)alloc(16384 * 2);
        blkWS[i] = (_Float16*)alloc(32768 * 2);
    }
    _Float16* w_inc1 = (_Float16*)alloc(92160L * 2);
    _Float16* w_inc2 = (_Float16*)alloc(36864L * 2);
    _Float16* w_d1a  = (_Float16*)alloc(73728L * 2);
    _Float16* w_d1b  = (_Float16*)alloc(147456L * 2);
    _Float16* w_d2a  = (_Float16*)alloc(294912L * 2);
    _Float16* w_d2b  = (_Float16*)alloc(589824L * 2);
    _Float16* w_u1c  = (_Float16*)alloc(32768L * 2);
    _Float16* w_u1a  = (_Float16*)alloc(294912L * 2);
    _Float16* w_u1b  = (_Float16*)alloc(147456L * 2);
    _Float16* w_u2c  = (_Float16*)alloc(8192L * 2);
    _Float16* w_u2a  = (_Float16*)alloc(73728L * 2);
    _Float16* w_u2b  = (_Float16*)alloc(36864L * 2);
    _Float16* w_outc = (_Float16*)alloc(8192L * 2);

    int*   fmaxb = (int*)alloc(65536L * 128 * 4);
    int*   dminb = (int*)alloc(65536L * 4);
    int*   dmaxb = (int*)alloc(65536L * 4);
    float* dsumb = (float*)alloc(65536L * 4);
    float* cntb  = (float*)alloc(65536L * 4);

    _Float16* x0buf = (_Float16*)alloc(65536L * 160 * 2);
    _Float16* t64a  = (_Float16*)alloc(16L * 4096 * 64 * 2);
    _Float16* cat2  = (_Float16*)alloc(16L * 4096 * 128 * 2);
    _Float16* p1b   = (_Float16*)alloc(16L * 1024 * 64 * 2);
    _Float16* t128a = (_Float16*)alloc(16L * 1024 * 128 * 2);
    _Float16* cat1  = (_Float16*)alloc(16L * 1024 * 256 * 2);
    _Float16* p2b   = (_Float16*)alloc(16L * 256 * 128 * 2);
    _Float16* t256a = (_Float16*)alloc(16L * 256 * 256 * 2);
    _Float16* t256b = (_Float16*)alloc(16L * 256 * 256 * 2);
    _Float16* t128b = (_Float16*)alloc(16L * 1024 * 128 * 2);
    _Float16* t128c = (_Float16*)alloc(16L * 1024 * 128 * 2);
    _Float16* t64b  = (_Float16*)alloc(16L * 4096 * 64 * 2);
    _Float16* t64c  = (_Float16*)alloc(16L * 4096 * 64 * 2);

    // ---- weight swizzles ----
    for (int i = 0; i < 5; ++i) {
        swz_mat<<<nb256(32768), 256, 0, stream>>>((const float*)d_in[3 + 5 * i + 0], 256, 128, blkW0[i]);
        swz_mat<<<nb256(16384), 256, 0, stream>>>((const float*)d_in[3 + 5 * i + 2], 128, 128, blkW1[i]);
        swz_mat<<<nb256(32768), 256, 0, stream>>>((const float*)d_in[3 + 5 * i + 4], 256, 128, blkWS[i]);
    }
    swz_conv<<<nb256(92160L), 256, 0, stream>>>((const float*)d_in[28], 64, 131, 3, 160, w_inc1);
    swz_conv<<<nb256(36864L), 256, 0, stream>>>((const float*)d_in[30], 64, 64, 3, 64, w_inc2);
    swz_conv<<<nb256(73728L), 256, 0, stream>>>((const float*)d_in[32], 128, 64, 3, 64, w_d1a);
    swz_conv<<<nb256(147456L), 256, 0, stream>>>((const float*)d_in[34], 128, 128, 3, 128, w_d1b);
    swz_conv<<<nb256(294912L), 256, 0, stream>>>((const float*)d_in[36], 256, 128, 3, 128, w_d2a);
    swz_conv<<<nb256(589824L), 256, 0, stream>>>((const float*)d_in[38], 256, 256, 3, 256, w_d2b);
    swz_conv<<<nb256(32768L), 256, 0, stream>>>((const float*)d_in[40], 128, 256, 1, 256, w_u1c);
    swz_conv<<<nb256(294912L), 256, 0, stream>>>((const float*)d_in[42], 128, 256, 3, 256, w_u1a);
    swz_conv<<<nb256(147456L), 256, 0, stream>>>((const float*)d_in[44], 128, 128, 3, 128, w_u1b);
    swz_conv<<<nb256(8192L), 256, 0, stream>>>((const float*)d_in[46], 64, 128, 1, 128, w_u2c);
    swz_conv<<<nb256(73728L), 256, 0, stream>>>((const float*)d_in[48], 64, 128, 3, 128, w_u2a);
    swz_conv<<<nb256(36864L), 256, 0, stream>>>((const float*)d_in[50], 64, 64, 3, 64, w_u2b);
    swz_conv<<<nb256(8192L), 256, 0, stream>>>((const float*)d_in[52], 128, 64, 1, 64, w_outc);

    // ---- point MLP ----
    fcpos_k<<<nb256(NPTS * 32), 256, 0, stream>>>(p, fcw, fcb, net256, NPTS);
    resblock_wmma<<<1024, 256, 0, stream>>>(net256, 256, net256 + 128, 256, 0, T,
                                            blkW0[0], (const float*)d_in[4],
                                            blkW1[0], (const float*)d_in[6], blkWS[0], netA);
    for (int i = 1; i < 5; ++i) {
        _Float16* cur = (i & 1) ? netA : netB;
        _Float16* nxt = (i & 1) ? netB : netA;
        poolmax_k<<<512, 256, 0, stream>>>(cur, pooled, T);
        resblock_wmma<<<1024, 256, 0, stream>>>(cur, 128, pooled, 0, 1, T,
                                                blkW0[i], (const float*)d_in[3 + 5 * i + 1],
                                                blkW1[i], (const float*)d_in[3 + 5 * i + 3],
                                                blkWS[i], nxt);
    }
    _Float16* netF = netA;  // block4 output

    // ---- spherical scatter ----
    hipMemsetAsync(fmaxb, 0, 65536L * 128 * 4, stream);
    hipMemsetAsync(dminb, 0, 65536L * 4, stream);
    hipMemsetAsync(dmaxb, 0, 65536L * 4, stream);
    hipMemsetAsync(dsumb, 0, 65536L * 4, stream);
    hipMemsetAsync(cntb, 0, 65536L * 4, stream);
    scatter_k<<<nb256(NPTS * 4), 256, 0, stream>>>(p, netF, T, B, fmaxb, dminb, dmaxb, dsumb, cntb);
    finalize_k<<<nb256(65536L * 160), 256, 0, stream>>>(fmaxb, dminb, dmaxb, dsumb, cntb, x0buf);

    // ---- UNet (NHWC f16, implicit-GEMM WMMA convs) ----
    conv3x3_wmma<<<cgrid(16L * 64 * 4 * 4), 256, 0, stream>>>(x0buf, 64, 64, 160, 160, w_inc1,
        (const float*)d_in[29], t64a, 64, 0, 64, 16, 1);
    conv3x3_wmma<<<cgrid(16L * 64 * 4 * 4), 256, 0, stream>>>(t64a, 64, 64, 64, 64, w_inc2,
        (const float*)d_in[31], cat2, 128, 64, 64, 16, 1);                     // x1 -> cat2[ch 64..128)
    maxpool2_k<<<nb256(16L * 32 * 32 * 64), 256, 0, stream>>>(cat2, 64, 64, 128, 64, 64, p1b, 64, 16);
    conv3x3_wmma<<<cgrid(16L * 32 * 2 * 8), 256, 0, stream>>>(p1b, 32, 32, 64, 64, w_d1a,
        (const float*)d_in[33], t128a, 128, 0, 128, 16, 1);
    conv3x3_wmma<<<cgrid(16L * 32 * 2 * 8), 256, 0, stream>>>(t128a, 32, 32, 128, 128, w_d1b,
        (const float*)d_in[35], cat1, 256, 128, 128, 16, 1);                   // x2 -> cat1[ch 128..256)
    maxpool2_k<<<nb256(16L * 16 * 16 * 128), 256, 0, stream>>>(cat1, 32, 32, 256, 128, 128, p2b, 128, 16);
    conv3x3_wmma<<<cgrid(16L * 16 * 1 * 16), 256, 0, stream>>>(p2b, 16, 16, 128, 128, w_d2a,
        (const float*)d_in[37], t256a, 256, 0, 256, 16, 1);
    conv3x3_wmma<<<cgrid(16L * 16 * 1 * 16), 256, 0, stream>>>(t256a, 16, 16, 256, 256, w_d2b,
        (const float*)d_in[39], t256b, 256, 0, 256, 16, 1);                    // x3
    conv1x1_wmma<<<cgrid(16L * 32 * 2 * 8), 256, 0, stream>>>(t256b, 16, 16, 256, 256, w_u1c,
        (const float*)d_in[41], cat1, nullptr, 32, 32, 256, 128, 16, 0, 1);    // up+1x1 -> cat1[ch 0..128)
    conv3x3_wmma<<<cgrid(16L * 32 * 2 * 8), 256, 0, stream>>>(cat1, 32, 32, 256, 256, w_u1a,
        (const float*)d_in[43], t128b, 128, 0, 128, 16, 1);
    conv3x3_wmma<<<cgrid(16L * 32 * 2 * 8), 256, 0, stream>>>(t128b, 32, 32, 128, 128, w_u1b,
        (const float*)d_in[45], t128c, 128, 0, 128, 16, 1);
    conv1x1_wmma<<<cgrid(16L * 64 * 4 * 4), 256, 0, stream>>>(t128c, 32, 32, 128, 128, w_u2c,
        (const float*)d_in[47], cat2, nullptr, 64, 64, 128, 64, 16, 0, 1);     // up+1x1 -> cat2[ch 0..64)
    conv3x3_wmma<<<cgrid(16L * 64 * 4 * 4), 256, 0, stream>>>(cat2, 64, 64, 128, 128, w_u2a,
        (const float*)d_in[49], t64b, 64, 0, 64, 16, 1);
    conv3x3_wmma<<<cgrid(16L * 64 * 4 * 4), 256, 0, stream>>>(t64b, 64, 64, 64, 64, w_u2b,
        (const float*)d_in[51], t64c, 64, 0, 64, 16, 1);
    conv1x1_wmma<<<cgrid(16L * 64 * 4 * 8), 256, 0, stream>>>(t64c, 64, 64, 64, 64, w_outc,
        (const float*)d_in[53], nullptr, (float*)d_out, 64, 64, 128, 128, 16, 0, 0);

    centers_k<<<1, 64, 0, stream>>>((float*)d_out + 8388608L);
}
